// UTPM_28673201668371
// MI455X (gfx1250) — matrix operand order repaired and checked by reading, not compiled, not run on hardware
//
#include <hip/hip_runtime.h>

// Problem constants (match reference)
#define Bn 2048
#define Sn 50
#define En 300
#define Tn 8
#define Un 300
#define Ln 1000
#define NPAIRn 44850
#define K7PAD 44864      // NPAIR padded to multiple of 32 (bf16 WMMA K-step)
#define N7ROWS 320       // E=300 padded so every n-tile row exists (5 blocks * 64)
#define EPSf 1e-5f

typedef __attribute__((ext_vector_type(2)))  float  v2f;
typedef __attribute__((ext_vector_type(8)))  float  v8f;
typedef __attribute__((ext_vector_type(16))) __bf16 v16bf;

// ---------------------------------------------------------------------------
// K0: pair table  p -> (i, j), g[p] = v_i . v_j   (row-major upper triangle)
// ---------------------------------------------------------------------------
__device__ __forceinline__ int pair_base(int i) { return i * (2 * En - i - 1) / 2; }

__global__ void k_pairs(const float* __restrict__ v, int* __restrict__ iu,
                        int* __restrict__ ju, float* __restrict__ g) {
    int p = blockIdx.x * 256 + threadIdx.x;
    if (p >= NPAIRn) return;
    float disc = (2.f * En - 1.f) * (2.f * En - 1.f) - 8.f * (float)p;
    int i = (int)((2.f * En - 1.f - sqrtf(disc)) * 0.5f);
    if (i < 0) i = 0;
    if (i > En - 2) i = En - 2;
    while (i > 0 && p < pair_base(i)) --i;
    while (p >= pair_base(i + 1)) ++i;
    int j = i + 1 + (p - pair_base(i));
    iu[p] = i; ju[p] = j;
    float acc = 0.f;
#pragma unroll
    for (int c = 0; c < 4; ++c) acc += v[i * 4 + c] * v[j * 4 + c];
    g[p] = acc;
}

// ---------------------------------------------------------------------------
// K1: fc7_w [E, NPAIR] f32 -> bf16, padded [N7ROWS, K7PAD] (zeros in pad)
// ---------------------------------------------------------------------------
__global__ void k_w7cvt(const float* __restrict__ w, __bf16* __restrict__ o) {
    size_t idx = (size_t)blockIdx.x * 256 + threadIdx.x;
    if (idx >= (size_t)N7ROWS * K7PAD) return;
    int n = (int)(idx / K7PAD);
    int k = (int)(idx % K7PAD);
    float val = (n < En && k < NPAIRn) ? w[(size_t)n * NPAIRn + k] : 0.f;
    o[idx] = (__bf16)val;
}

// ---------------------------------------------------------------------------
// K2: z{1,2}[b,s,t] = relu(x[b,s,:] . w[t,:] + b[t]);  sign[b,s]
//     one 64-thread block per (b,s) row
// ---------------------------------------------------------------------------
__global__ void k_attn_z(const float* __restrict__ x,
                         const float* __restrict__ w1, const float* __restrict__ b1,
                         const float* __restrict__ w2, const float* __restrict__ b2,
                         float* __restrict__ z1, float* __restrict__ z2,
                         float* __restrict__ sgn) {
    int bs = blockIdx.x;
    const float* xr = x + (size_t)bs * En;
    int tid = threadIdx.x;
    float p1[Tn] = {}, p2[Tn] = {};
    float ab = 0.f;
    for (int e = tid; e < En; e += 64) {
        float xe = xr[e];
        ab += fabsf(xe);
#pragma unroll
        for (int t = 0; t < Tn; ++t) {
            p1[t] += xe * w1[t * En + e];
            p2[t] += xe * w2[t * En + e];
        }
    }
#pragma unroll
    for (int off = 16; off; off >>= 1) {
        ab += __shfl_xor(ab, off, 32);
#pragma unroll
        for (int t = 0; t < Tn; ++t) {
            p1[t] += __shfl_xor(p1[t], off, 32);
            p2[t] += __shfl_xor(p2[t], off, 32);
        }
    }
    __shared__ float sh[2][17];
    if ((tid & 31) == 0) {
        int w = tid >> 5;
        sh[w][0] = ab;
#pragma unroll
        for (int t = 0; t < Tn; ++t) { sh[w][1 + t] = p1[t]; sh[w][9 + t] = p2[t]; }
    }
    __syncthreads();
    if (tid < Tn) {
        float r = sh[0][1 + tid] + sh[1][1 + tid] + b1[tid];
        z1[(size_t)bs * Tn + tid] = r > 0.f ? r : 0.f;
    } else if (tid < 2 * Tn) {
        int t = tid - Tn;
        float r = sh[0][9 + t] + sh[1][9 + t] + b2[t];
        z2[(size_t)bs * Tn + t] = r > 0.f ? r : 0.f;
    } else if (tid == 2 * Tn) {
        float s = sh[0][0] + sh[1][0];
        sgn[bs] = (s > 0.f) ? 1.f : 0.f;
    }
}

// ---------------------------------------------------------------------------
// K3: BN stats per (head, s) over (B, T): mean + rsqrt(var+eps)
// ---------------------------------------------------------------------------
__global__ void k_bnstats_s(const float* __restrict__ z1, const float* __restrict__ z2,
                            float* __restrict__ m1, float* __restrict__ r1,
                            float* __restrict__ m2, float* __restrict__ r2) {
    int head = blockIdx.x / Sn;
    int s = blockIdx.x % Sn;
    const float* z = head ? z2 : z1;
    float sum = 0.f, sum2 = 0.f;
    for (int i = threadIdx.x; i < Bn * Tn; i += 256) {
        int b = i >> 3, t = i & 7;
        float v = z[(size_t)(b * Sn + s) * Tn + t];
        sum += v; sum2 += v * v;
    }
    __shared__ float rs[256], rs2[256];
    rs[threadIdx.x] = sum; rs2[threadIdx.x] = sum2;
    __syncthreads();
    for (int off = 128; off; off >>= 1) {
        if (threadIdx.x < off) {
            rs[threadIdx.x]  += rs[threadIdx.x + off];
            rs2[threadIdx.x] += rs2[threadIdx.x + off];
        }
        __syncthreads();
    }
    if (threadIdx.x == 0) {
        const float inv = 1.f / (float)(Bn * Tn);
        float m = rs[0] * inv;
        float var = rs2[0] * inv - m * m;
        if (head) { m2[s] = m; r2[s] = rsqrtf(var + EPSf); }
        else      { m1[s] = m; r1[s] = rsqrtf(var + EPSf); }
    }
}

// ---------------------------------------------------------------------------
// K4: scores -> mask -> softmax over S -> pooled h[b, 0:600]
//     one 320-thread block per b
// ---------------------------------------------------------------------------
__global__ void k_attn_pool(const float* __restrict__ x,
                            const float* __restrict__ z1, const float* __restrict__ z2,
                            const float* __restrict__ m1, const float* __restrict__ r1,
                            const float* __restrict__ m2, const float* __restrict__ r2,
                            const float* __restrict__ q1, const float* __restrict__ q2,
                            const float* __restrict__ sgn, float* __restrict__ h) {
    int b = blockIdx.x;
    int tid = threadIdx.x;
    __shared__ float w1[Sn], w2[Sn];
    const float PADV = -4294967295.0f;  // -2^32 + 1
    if (tid < Sn) {
        int s = tid;
        float acc = 0.f;
#pragma unroll
        for (int t = 0; t < Tn; ++t)
            acc += (z1[(size_t)(b * Sn + s) * Tn + t] - m1[s]) * r1[s] * q1[t];
        w1[s] = (sgn[b * Sn + s] == 0.f) ? PADV : acc;
    } else if (tid >= 64 && tid < 64 + Sn) {
        int s = tid - 64;
        float acc = 0.f;
#pragma unroll
        for (int t = 0; t < Tn; ++t)
            acc += (z2[(size_t)(b * Sn + s) * Tn + t] - m2[s]) * r2[s] * q2[t];
        w2[s] = (sgn[b * Sn + s] == 0.f) ? PADV : acc;
    }
    __syncthreads();
    if (tid < 2) {
        float* ws = tid ? w2 : w1;
        float mx = -3.4e38f;
        for (int s = 0; s < Sn; ++s) mx = fmaxf(mx, ws[s]);
        float sum = 0.f;
        for (int s = 0; s < Sn; ++s) { float e = expf(ws[s] - mx); ws[s] = e; sum += e; }
        float inv = 1.f / sum;
        for (int s = 0; s < Sn; ++s) ws[s] *= inv;
    }
    __syncthreads();
    if (tid < En) {
        float a1 = 0.f, a2 = 0.f;
        for (int s = 0; s < Sn; ++s) {
            float xv = x[((size_t)b * Sn + s) * En + tid];
            a1 += w1[s] * xv; a2 += w2[s] * xv;
        }
        h[(size_t)b * (2 * En) + tid] = a1;
        h[(size_t)b * (2 * En) + En + tid] = a2;
    }
}

// ---------------------------------------------------------------------------
// Generic f32 WMMA GEMM:  C[M,N] = A[M,K] @ W[N,K]^T (+bias)
// v_wmma_f32_16x16x4_f32; 4 waves/block, each owns a 16x16 tile; M % 64 == 0.
// Full n-tiles take a fast loop (loads feed WMMA directly); the single edge
// tile uses branchless clamp + 0/1-mask multiply (no EXEC manipulation).
// ---------------------------------------------------------------------------
__global__ void k_gemm_f32(const float* __restrict__ A, int lda,
                           const float* __restrict__ W, int ldw,
                           const float* __restrict__ bias,
                           float* __restrict__ C, int ldc,
                           int N, int K) {
    int wave = threadIdx.x >> 5;
    int lane = threadIdx.x & 31;
    int m0 = blockIdx.y * 64 + wave * 16;
    int n0 = blockIdx.x * 16;
    int lm = lane & 15;
    int ksel = (lane & 16) ? 2 : 0;     // A/B VGPR0 holds K0|K2, VGPR1 holds K1|K3
    const float* Ap = A + (size_t)(m0 + lm) * lda + ksel;
    int brow = n0 + lm;
    v8f acc = {};
    if (n0 + 16 <= N) {
        // ---- fast path: full tile, unmasked loads straight into WMMA ----
        const float* Wp = W + (size_t)brow * ldw + ksel;
        for (int kb = 0; kb < K; kb += 4) {
            v2f a, bq;
            a.x = Ap[kb];
            a.y = Ap[kb + 1];
            bq.x = Wp[kb];
            bq.y = Wp[kb + 1];
            acc = __builtin_amdgcn_wmma_f32_16x16x4_f32(false, a, false, bq,
                                                        (short)0, acc, false, false);
        }
    } else {
        // ---- edge tile: clamp row, zero via multiply (branchless) ----
        float wmask = (brow < N) ? 1.f : 0.f;
        int brow_c = (brow < N) ? brow : (N - 1);
        const float* Wp = W + (size_t)brow_c * ldw + ksel;
        for (int kb = 0; kb < K; kb += 4) {
            v2f a, bq;
            a.x = Ap[kb];
            a.y = Ap[kb + 1];
            bq.x = Wp[kb] * wmask;
            bq.y = Wp[kb + 1] * wmask;
            acc = __builtin_amdgcn_wmma_f32_16x16x4_f32(false, a, false, bq,
                                                        (short)0, acc, false, false);
        }
    }
    if (brow < N) {
        float bv = bias ? bias[brow] : 0.f;
        int mb = m0 + ((lane & 16) ? 8 : 0);
#pragma unroll
        for (int r = 0; r < 8; ++r)
            C[(size_t)(mb + r) * ldc + brow] = acc[r] + bv;
    }
}

// ---------------------------------------------------------------------------
// Per-column mean / rstd over B rows
// ---------------------------------------------------------------------------
__global__ void k_colstats(const float* __restrict__ in, int ncols,
                           float* __restrict__ mean, float* __restrict__ rstd) {
    int col = blockIdx.x;
    float s = 0.f, s2 = 0.f;
    for (int b = threadIdx.x; b < Bn; b += 256) {
        float v = in[(size_t)b * ncols + col];
        s += v; s2 += v * v;
    }
    __shared__ float rs[256], rs2[256];
    rs[threadIdx.x] = s; rs2[threadIdx.x] = s2;
    __syncthreads();
    for (int off = 128; off; off >>= 1) {
        if (threadIdx.x < off) {
            rs[threadIdx.x]  += rs[threadIdx.x + off];
            rs2[threadIdx.x] += rs2[threadIdx.x + off];
        }
        __syncthreads();
    }
    if (threadIdx.x == 0) {
        float m = rs[0] / (float)Bn;
        float var = rs2[0] / (float)Bn - m * m;
        mean[col] = m; rstd[col] = rsqrtf(var + EPSf);
    }
}

__global__ void k_bnapply(const float* __restrict__ in,
                          const float* __restrict__ mean, const float* __restrict__ rstd,
                          float* __restrict__ out, int ostride, int ooff) {
    int idx = blockIdx.x * 256 + threadIdx.x;
    if (idx >= Bn * En) return;
    int b = idx / En, c = idx - b * En;
    out[(size_t)b * ostride + ooff + c] = (in[idx] - mean[c]) * rstd[c];
}

// ---------------------------------------------------------------------------
// Fused CrossFeature + fc7 GEMM (the 55-GFLOP stage), bf16 WMMA 16x16x32.
//   h7pre[b,n] = fc7_b[n] + sum_p (h5[b,iu]*h5[b,ju]*g[p]) * w7[n,p]
// Block: 256 thr (8 waves) -> 32(M) x 64(N) tile; h5 rows cached in LDS.
// B-fragment is a contiguous 32B bf16 load; A-fragment is built in-register.
// ---------------------------------------------------------------------------
__global__ void k_fc7_fused(const float* __restrict__ h5,
                            const __bf16* __restrict__ w7,
                            const int* __restrict__ iu, const int* __restrict__ ju,
                            const float* __restrict__ g,
                            const float* __restrict__ bias,
                            float* __restrict__ out) {
    __shared__ float lh5[32 * En];      // 37.5 KB of the WGP's 320 KB LDS
    int tid = threadIdx.x;
    int mblk = blockIdx.y * 32;
    for (int idx = tid; idx < 32 * En; idx += 256) {
        int r = idx / En, c = idx - r * En;
        lh5[idx] = h5[(size_t)(mblk + r) * En + c];
    }
    __syncthreads();
    int wave = tid >> 5, lane = tid & 31;
    int m0 = (wave & 1) * 16;                 // LDS-local m tile
    int n0 = blockIdx.x * 64 + (wave >> 1) * 16;
    int lm = lane & 15;
    int n = n0 + lm;                          // < N7ROWS always (pad rows are zero)
    const float* hrow = lh5 + (m0 + lm) * En;
    int kA0 = (lane & 16) ? 16 : 0;           // A: contiguous k = 16h + s (matches B)
    int kB0 = (lane & 16) ? 16 : 0;           // B 32x16 bf16 layout
    const __bf16* wp = w7 + (size_t)n * K7PAD + kB0;
    v8f acc = {};
    for (int kb = 0; kb < K7PAD; kb += 32) {
        __builtin_prefetch((const void*)(wp + kb + 2048), 0, 0);
        v16bf bfrag = *(const v16bf*)(wp + kb);   // contiguous: frag elem s <-> k = kB0+s
        v16bf afrag;
#pragma unroll
        for (int d = 0; d < 8; ++d) {
            int kdb = kA0 + 2 * d;
#pragma unroll
            for (int par = 0; par < 2; ++par) {
                int p = kb + kdb + par;
                float vv = 0.f;
                if (p < NPAIRn)
                    vv = hrow[iu[p]] * hrow[ju[p]] * g[p];
                afrag[2 * d + par] = (__bf16)vv;
            }
        }
        acc = __builtin_amdgcn_wmma_f32_16x16x32_bf16(false, afrag, false, bfrag,
                                                      (short)0, acc, false, false);
    }
    if (n < En) {
        int mb = mblk + m0 + ((lane & 16) ? 8 : 0);
        float bv = bias[n];
#pragma unroll
        for (int r = 0; r < 8; ++r)
            out[(size_t)(mb + r) * En + n] = acc[r] + bv;
    }
}

// ---------------------------------------------------------------------------
extern "C" void kernel_launch(void* const* d_in, const int* in_sizes, int n_in,
                              void* d_out, int out_size, void* d_ws, size_t ws_size,
                              hipStream_t stream) {
    (void)in_sizes; (void)n_in; (void)out_size; (void)ws_size;
    const float* x    = (const float*)d_in[0];
    const float* lab  = (const float*)d_in[1];
    const float* fc1w = (const float*)d_in[2];
    const float* fc1b = (const float*)d_in[3];
    const float* q1   = (const float*)d_in[4];
    const float* fc3w = (const float*)d_in[5];
    const float* fc3b = (const float*)d_in[6];
    const float* q2   = (const float*)d_in[7];
    const float* fc5w = (const float*)d_in[8];
    const float* fc5b = (const float*)d_in[9];
    const float* fc6w = (const float*)d_in[10];
    const float* fc6b = (const float*)d_in[11];
    const float* fc7w = (const float*)d_in[12];
    const float* fc7b = (const float*)d_in[13];
    const float* fc8w = (const float*)d_in[14];
    const float* fc8b = (const float*)d_in[15];
    const float* vmat = (const float*)d_in[16];
    float* out = (float*)d_out;

    char* wsp = (char*)d_ws;
    auto take = [&](size_t bytes) -> char* {
        char* r = wsp;
        wsp += (bytes + 255) & ~(size_t)255;
        return r;
    };
    float* z1    = (float*)take((size_t)Bn * Sn * Tn * 4);
    float* z2    = (float*)take((size_t)Bn * Sn * Tn * 4);
    float* sgn   = (float*)take((size_t)Bn * Sn * 4);
    float* m1    = (float*)take(Sn * 4);
    float* r1    = (float*)take(Sn * 4);
    float* m2    = (float*)take(Sn * 4);
    float* r2    = (float*)take(Sn * 4);
    float* h     = (float*)take((size_t)Bn * 2 * En * 4);
    float* h5pre = (float*)take((size_t)Bn * En * 4);
    float* h5    = (float*)take((size_t)Bn * En * 4);
    float* h6pre = (float*)take((size_t)Bn * En * 4);
    float* h7pre = (float*)take((size_t)Bn * En * 4);
    float* hc    = (float*)take((size_t)Bn * 2 * En * 4);
    float* out8  = (float*)take((size_t)Bn * En * 4);
    float* cmean = (float*)take(En * 4);
    float* crstd = (float*)take(En * 4);
    int*   iu    = (int*)take((size_t)NPAIRn * 4);
    int*   ju    = (int*)take((size_t)NPAIRn * 4);
    float* g     = (float*)take((size_t)NPAIRn * 4);
    __bf16* w7bf = (__bf16*)take((size_t)N7ROWS * K7PAD * 2);

    // Stage 0: pair tables + weight conversion
    k_pairs<<<(NPAIRn + 255) / 256, 256, 0, stream>>>(vmat, iu, ju, g);
    {
        size_t tot = (size_t)N7ROWS * K7PAD;
        k_w7cvt<<<(unsigned)((tot + 255) / 256), 256, 0, stream>>>(fc7w, w7bf);
    }
    // Stage 1: attention heads
    k_attn_z<<<Bn * Sn, 64, 0, stream>>>(x, fc1w, fc1b, fc3w, fc3b, z1, z2, sgn);
    k_bnstats_s<<<2 * Sn, 256, 0, stream>>>(z1, z2, m1, r1, m2, r2);
    k_attn_pool<<<Bn, 320, 0, stream>>>(x, z1, z2, m1, r1, m2, r2, q1, q2, sgn, h);
    // Stage 2: fc5 -> BN -> h5
    k_gemm_f32<<<dim3((En + 15) / 16, Bn / 64), 128, 0, stream>>>(
        h, 2 * En, fc5w, 2 * En, fc5b, h5pre, En, En, 2 * En);
    k_colstats<<<En, 256, 0, stream>>>(h5pre, En, cmean, crstd);
    k_bnapply<<<(Bn * En + 255) / 256, 256, 0, stream>>>(h5pre, cmean, crstd, h5, En, 0);
    // Stage 3: fc6 -> BN -> hc[:, 0:300]
    k_gemm_f32<<<dim3((En + 15) / 16, Bn / 64), 128, 0, stream>>>(
        h5, En, fc6w, En, fc6b, h6pre, En, En, En);
    k_colstats<<<En, 256, 0, stream>>>(h6pre, En, cmean, crstd);
    k_bnapply<<<(Bn * En + 255) / 256, 256, 0, stream>>>(h6pre, cmean, crstd, hc, 2 * En, 0);
    // Stage 4: fused cross-feature + fc7 (bf16 WMMA) -> BN -> hc[:, 300:600]
    k_fc7_fused<<<dim3(N7ROWS / 64, Bn / 32), 256, 0, stream>>>(
        h5, w7bf, iu, ju, g, fc7b, h7pre);
    k_colstats<<<En, 256, 0, stream>>>(h7pre, En, cmean, crstd);
    k_bnapply<<<(Bn * En + 255) / 256, 256, 0, stream>>>(h7pre, cmean, crstd, hc, 2 * En, En);
    // Stage 5: fc8 then label projection
    k_gemm_f32<<<dim3((Un + 15) / 16, Bn / 64), 128, 0, stream>>>(
        hc, 2 * En, fc8w, 2 * En, fc8b, out8, Un, Un, 2 * En);
    k_gemm_f32<<<dim3((Ln + 15) / 16, Bn / 64), 128, 0, stream>>>(
        out8, Un, lab, Un, (const float*)nullptr, out, Ln, Ln, Un);
}
